// DeeperGCN_38680475467995
// MI455X (gfx1250) — matrix operand order repaired and checked by reading, not compiled, not run on hardware
//
#include <hip/hip_runtime.h>

typedef __attribute__((ext_vector_type(16))) _Float16 v16h;
typedef __attribute__((ext_vector_type(8)))  _Float16 v8h;
typedef __attribute__((ext_vector_type(8)))  float    v8f;

#define N_IN   128
#define HID    64
#define HID2   128
#define NOUTC  112
#define NLAYERS 7

#define DEV static __device__ __forceinline__

// ---------------- cross-lane reductions (wave32) ----------------
DEV float rsum16(float v) {   // sum within each 16-lane half-group
    v += __shfl_xor(v, 1);
    v += __shfl_xor(v, 2);
    v += __shfl_xor(v, 4);
    v += __shfl_xor(v, 8);
    return v;
}
DEV float rsum32(float v) {   // sum over the full wave32
    v = rsum16(v);
    v += __shfl_xor(v, 16);
    return v;
}

// float atomic-max via monotonic int encoding (positive: int max, negative: uint min)
DEV void atomicMaxF(float* a, float v) {
    int iv = __float_as_int(v);
    if (iv >= 0) atomicMax((int*)a, iv);
    else         atomicMin((unsigned int*)a, (unsigned int)iv);
}

// ---------------- WMMA fragment loads (contiguous k = 16*h + e for A and B) ----------------
// A (16x32 f16, row-major, lda halfwords): lane half h, m = lane&15
//   e in 0..7  -> K = k0 + 16*h + e          (contiguous 16B chunk)
//   e in 8..15 -> K = k0 + 16*h + e          (second contiguous 16B chunk)
DEV v16h load_a_frag(const _Float16* A, int lda, int k0, int lane) {
    int hf = lane >> 4, m = lane & 15;
    const _Float16* p = A + (size_t)m * lda + k0 + 16 * hf;
    v8h lo = *(const v8h*)p;
    v8h hi = *(const v8h*)(p + 8);
    v16h a;
#pragma unroll
    for (int e = 0; e < 8; ++e) { a[e] = lo[e]; a[8 + e] = hi[e]; }
    return a;
}

// B fragment from pre-swizzled weights: one contiguous 32-byte load per lane.
// Wsw block layout: [(kt*NT + nt)][lane][e], element e holds K = kt*32 + 16*(lane>>4) + e,
// column n = nt*16 + (lane&15).
DEV v16h load_b_frag_sw(const _Float16* Wsw, int kt, int nt, int NT, int lane) {
    const _Float16* p = Wsw + (((size_t)kt * NT + nt) * 32 + lane) * 16;
    return *(const v16h*)p;
}

// ---------------- generic 16-row-block WMMA GEMM ----------------
// One wave computes rows [row0, row0+16) x all NT*16 output columns.
// MODE 0: out_f32 = A@W + bias (+ residual in-place)
// MODE 1: out_f16 = relu(LayerNorm(A@W + bias) * g + b)   (LN across NT*16 cols)
template <int KD32, int NT, int MODE>
__global__ void __launch_bounds__(256)
gemm_rowblock(const _Float16* __restrict__ A,     // [nrows x KD32*32]
              const _Float16* __restrict__ Wsw,   // swizzled [KD32*NT][32][16]
              const float* __restrict__ bias,     // [NT*16]
              const float* __restrict__ lng,      // MODE 1 only
              const float* __restrict__ lnb,      // MODE 1 only
              float* __restrict__ outf,           // MODE 0
              _Float16* __restrict__ outh,        // MODE 1
              int nrows, int addres)
{
    const int K  = KD32 * 32;
    const int NO = NT * 16;
    int wid  = (int)((blockIdx.x * blockDim.x + threadIdx.x) >> 5);
    int lane = threadIdx.x & 31;
    int row0 = wid * 16;
    if (row0 >= nrows) return;
    int hf = lane >> 4, nn = lane & 15;

    const _Float16* Ar = A + (size_t)row0 * K;
    v16h af[KD32];
#pragma unroll
    for (int k = 0; k < KD32; ++k) af[k] = load_a_frag(Ar, K, k * 32, lane);

    v8f acc[NT];
#pragma unroll
    for (int t = 0; t < NT; ++t) {
        v8f c = {};
#pragma unroll
        for (int k = 0; k < KD32; ++k) {
            v16h bf = load_b_frag_sw(Wsw, k, t, NT, lane);
            c = __builtin_amdgcn_wmma_f32_16x16x32_f16(
                    false, af[k], false, bf, (short)0, c, false, false);
        }
        acc[t] = c;
    }

    // bias (same for both lane halves: column = t*16 + nn)
#pragma unroll
    for (int t = 0; t < NT; ++t) {
        float bv = bias[t * 16 + nn];
#pragma unroll
        for (int r = 0; r < 8; ++r) acc[t][r] += bv;
    }

    if (MODE == 0) {
#pragma unroll
        for (int t = 0; t < NT; ++t) {
            float* po = outf + (size_t)(row0 + 8 * hf) * NO + t * 16 + nn;
#pragma unroll
            for (int r = 0; r < 8; ++r) {       // row m = r + 8*hf
                float v = acc[t][r];
                if (addres) v += po[(size_t)r * NO];
                po[(size_t)r * NO] = v;
            }
        }
    } else {
        // LayerNorm over NO columns of each row; row m lives in component r
        // across the 16 lanes of this half-group and across the NT tiles.
        float mu[8], rs[8];
#pragma unroll
        for (int r = 0; r < 8; ++r) {
            float s = 0.f;
#pragma unroll
            for (int t = 0; t < NT; ++t) s += acc[t][r];
            mu[r] = rsum16(s) * (1.0f / NO);
        }
#pragma unroll
        for (int r = 0; r < 8; ++r) {
            float q = 0.f;
#pragma unroll
            for (int t = 0; t < NT; ++t) { float d = acc[t][r] - mu[r]; q += d * d; }
            rs[r] = rsqrtf(rsum16(q) * (1.0f / NO) + 1e-5f);
        }
#pragma unroll
        for (int t = 0; t < NT; ++t) {
            float gv = lng[t * 16 + nn];
            float bv = lnb[t * 16 + nn];
            _Float16* po = outh + (size_t)(row0 + 8 * hf) * NO + t * 16 + nn;
#pragma unroll
            for (int r = 0; r < 8; ++r) {
                float v = (acc[t][r] - mu[r]) * rs[r] * gv + bv;
                po[(size_t)r * NO] = (_Float16)fmaxf(v, 0.f);
            }
        }
    }
}

// ---------------- small helper kernels ----------------
__global__ void convert_kernel(const float* __restrict__ s, _Float16* __restrict__ d, int n) {
    int i = blockIdx.x * blockDim.x + threadIdx.x;
    if (i < n) d[i] = (_Float16)s[i];
}

// Swizzle a batch of row-major f32 weights [batch][K][NO] into f16 B-fragment
// layout [batch][kt*NT+nt][lane][e]:  K = kt*32 + 16*(lane>>4) + e, n = nt*16 + (lane&15).
__global__ void swizzle_w_kernel(const float* __restrict__ W, _Float16* __restrict__ Wsw,
                                 int K, int NO, int total /* = batch*K*NO */) {
    int i = blockIdx.x * blockDim.x + threadIdx.x;
    if (i >= total) return;
    int per  = K * NO;
    int b    = i / per;
    int o    = i - b * per;
    int NT   = NO >> 4;
    int e    = o & 15;
    int lane = (o >> 4) & 31;
    int tile = o >> 9;
    int nt   = tile % NT;
    int kt   = tile / NT;
    int hf   = lane >> 4, n = lane & 15;
    int k    = kt * 32 + 16 * hf + e;
    Wsw[i] = (_Float16)W[(size_t)b * per + (size_t)k * NO + nt * 16 + n];
}

__global__ void init_kernel(float* __restrict__ zmax, float* __restrict__ den,
                            float* __restrict__ num, int n) {
    int i = blockIdx.x * blockDim.x + threadIdx.x;
    if (i < n) { zmax[i] = -3.402823466e38f; den[i] = 0.f; num[i] = 0.f; }
}

// r = relu(LayerNorm(h)*g + b); one wave per row, 2 channels/lane. f32 + f16 outputs.
__global__ void ln_relu_kernel(const float* __restrict__ h, const float* __restrict__ g,
                               const float* __restrict__ b, float* __restrict__ r32,
                               _Float16* __restrict__ r16, int nrows)
{
    int wid  = (int)((blockIdx.x * blockDim.x + threadIdx.x) >> 5);
    int lane = threadIdx.x & 31;
    if (wid >= nrows) return;
    size_t base = (size_t)wid * HID;
    float x0 = h[base + lane], x1 = h[base + lane + 32];
    float mu = rsum32(x0 + x1) * (1.0f / HID);
    float d0 = x0 - mu, d1 = x1 - mu;
    float var = rsum32(d0 * d0 + d1 * d1) * (1.0f / HID);
    float rstd = rsqrtf(var + 1e-5f);
    float y0 = fmaxf(d0 * rstd * g[lane] + b[lane], 0.f);
    float y1 = fmaxf(d1 * rstd * g[lane + 32] + b[lane + 32], 0.f);
    r32[base + lane] = y0;           r32[base + lane + 32] = y1;
    r16[base + lane] = (_Float16)y0; r16[base + lane + 32] = (_Float16)y1;
}

// pass 1 over edges: per-dst per-channel max of z = t * (relu(x[src]) + eps)
__global__ void edge_max_kernel(const float* __restrict__ x, const int* __restrict__ ei,
                                const float* __restrict__ tarr, int layer,
                                float* __restrict__ zmax, int nedges)
{
    int wid  = (int)((blockIdx.x * blockDim.x + threadIdx.x) >> 5);
    int lane = threadIdx.x & 31;
    if (wid >= nedges) return;
    int s = ei[wid], d = ei[nedges + wid];
    float tv = tarr[layer];
    size_t sb = (size_t)s * HID, db = (size_t)d * HID;
    float m0 = fmaxf(x[sb + lane],      0.f) + 1e-7f;
    float m1 = fmaxf(x[sb + lane + 32], 0.f) + 1e-7f;
    atomicMaxF(&zmax[db + lane],      tv * m0);
    atomicMaxF(&zmax[db + lane + 32], tv * m1);
}

// pass 2 over edges: e = exp(z - zmax[dst]); den += e; num += m*e
__global__ void edge_sum_kernel(const float* __restrict__ x, const int* __restrict__ ei,
                                const float* __restrict__ tarr, int layer,
                                const float* __restrict__ zmax,
                                float* __restrict__ den, float* __restrict__ num, int nedges)
{
    int wid  = (int)((blockIdx.x * blockDim.x + threadIdx.x) >> 5);
    int lane = threadIdx.x & 31;
    if (wid >= nedges) return;
    int s = ei[wid], d = ei[nedges + wid];
    float tv = tarr[layer];
    size_t sb = (size_t)s * HID, db = (size_t)d * HID;
    float m0 = fmaxf(x[sb + lane],      0.f) + 1e-7f;
    float m1 = fmaxf(x[sb + lane + 32], 0.f) + 1e-7f;
    float e0 = __expf(tv * m0 - zmax[db + lane]);
    float e1 = __expf(tv * m1 - zmax[db + lane + 32]);
    atomicAdd(&den[db + lane],      e0);
    atomicAdd(&den[db + lane + 32], e1);
    atomicAdd(&num[db + lane],      m0 * e0);
    atomicAdd(&num[db + lane + 32], m1 * e1);
}

// per-node: agg = num/den; MsgNorm: hpre = x + scale * normalize(agg) * ||x||  -> f16
__global__ void combine_kernel(const float* __restrict__ x, const float* __restrict__ num,
                               const float* __restrict__ den, const float* __restrict__ msc,
                               int layer, _Float16* __restrict__ hpre, int nrows)
{
    int wid  = (int)((blockIdx.x * blockDim.x + threadIdx.x) >> 5);
    int lane = threadIdx.x & 31;
    if (wid >= nrows) return;
    size_t base = (size_t)wid * HID;
    float dn0 = den[base + lane],      dn1 = den[base + lane + 32];
    float a0  = num[base + lane]      / (dn0 > 0.f ? dn0 : 1.f);
    float a1  = num[base + lane + 32] / (dn1 > 0.f ? dn1 : 1.f);
    float na  = sqrtf(rsum32(a0 * a0 + a1 * a1));
    float x0  = x[base + lane], x1 = x[base + lane + 32];
    float nx  = sqrtf(rsum32(x0 * x0 + x1 * x1));
    float sc  = msc[layer];
    float inv = 1.f / fmaxf(na, 1e-12f);
    float h0  = x0 + sc * (a0 * inv) * nx;
    float h1  = x1 + sc * (a1 * inv) * nx;
    hpre[base + lane]      = (_Float16)h0;
    hpre[base + lane + 32] = (_Float16)h1;
}

// ---------------- driver ----------------
extern "C" void kernel_launch(void* const* d_in, const int* in_sizes, int n_in,
                              void* d_out, int out_size, void* d_ws, size_t ws_size,
                              hipStream_t stream)
{
    (void)n_in; (void)out_size; (void)ws_size;
    const float* x      = (const float*)d_in[0];
    const int*   ei     = (const int*)  d_in[1];
    const float* encW   = (const float*)d_in[2];
    const float* encb   = (const float*)d_in[3];
    const float* tarr   = (const float*)d_in[4];
    const float* mscale = (const float*)d_in[5];
    const float* W1     = (const float*)d_in[6];
    const float* b1     = (const float*)d_in[7];
    const float* g1     = (const float*)d_in[8];
    const float* bb1    = (const float*)d_in[9];
    const float* W2     = (const float*)d_in[10];
    const float* b2     = (const float*)d_in[11];
    const float* ng     = (const float*)d_in[12];
    const float* nb     = (const float*)d_in[13];
    const float* linW   = (const float*)d_in[14];
    const float* linb   = (const float*)d_in[15];

    const int N = in_sizes[0] / N_IN;
    const int E = in_sizes[1] / 2;
    float* out = (float*)d_out;

    // workspace carve-out (256B aligned slices)
    char* ws = (char*)d_ws;
    size_t off = 0;
    auto carve = [&](size_t bytes) -> char* {
        char* p = ws + off;
        off += (bytes + 255) & ~(size_t)255;
        return p;
    };
    _Float16* xh    = (_Float16*)carve((size_t)N * N_IN * 2);
    float*    h     = (float*)   carve((size_t)N * HID * 4);
    float*    r32   = (float*)   carve((size_t)N * HID * 4);
    _Float16* r16   = (_Float16*)carve((size_t)N * HID * 2);
    _Float16* hp16  = (_Float16*)carve((size_t)N * HID * 2);
    _Float16* u16   = (_Float16*)carve((size_t)N * HID2 * 2);
    float*    zmax  = (float*)   carve((size_t)N * HID * 4);
    float*    den   = (float*)   carve((size_t)N * HID * 4);
    float*    num   = (float*)   carve((size_t)N * HID * 4);
    _Float16* encWs = (_Float16*)carve((size_t)N_IN * HID * 2);
    _Float16* W1s   = (_Float16*)carve((size_t)NLAYERS * HID * HID2 * 2);
    _Float16* W2s   = (_Float16*)carve((size_t)NLAYERS * HID2 * HID * 2);
    _Float16* linWs = (_Float16*)carve((size_t)HID * NOUTC * 2);

    // activations f32 -> f16 (row-major; A fragments load contiguous 16B chunks)
    {
        int n = N * N_IN;
        convert_kernel<<<(n + 255) / 256, 256, 0, stream>>>(x, xh, n);
    }
    // weights f32 -> f16, pre-swizzled into B-fragment layout
    {
        int n = N_IN * HID;
        swizzle_w_kernel<<<(n + 255) / 256, 256, 0, stream>>>(encW, encWs, N_IN, HID, n);
    }
    {
        int n = NLAYERS * HID * HID2;
        swizzle_w_kernel<<<(n + 255) / 256, 256, 0, stream>>>(W1, W1s, HID, HID2, n);
    }
    {
        int n = NLAYERS * HID2 * HID;
        swizzle_w_kernel<<<(n + 255) / 256, 256, 0, stream>>>(W2, W2s, HID2, HID, n);
    }
    {
        int n = HID * NOUTC;
        swizzle_w_kernel<<<(n + 255) / 256, 256, 0, stream>>>(linW, linWs, HID, NOUTC, n);
    }

    const int gemm_blocks    = (int)((((N + 15) / 16) + 7) / 8);  // 8 waves/block, 16 rows/wave
    const int rowwave_blocks = (N + 7) / 8;                       // 1 wave per node row
    const int edge_blocks    = (E + 7) / 8;                       // 1 wave per edge
    const int initn          = N * HID;

    // encoder: h = x @ enc_W + enc_b
    gemm_rowblock<4, 4, 0><<<gemm_blocks, 256, 0, stream>>>(
        xh, encWs, encb, nullptr, nullptr, h, nullptr, N, 0);

    for (int l = 0; l < NLAYERS; ++l) {
        const float* xin = h;
        if (l > 0) {
            ln_relu_kernel<<<rowwave_blocks, 256, 0, stream>>>(
                h, ng + (size_t)l * HID, nb + (size_t)l * HID, r32, r16, N);
            xin = r32;
        }
        init_kernel<<<(initn + 255) / 256, 256, 0, stream>>>(zmax, den, num, initn);
        edge_max_kernel<<<edge_blocks, 256, 0, stream>>>(xin, ei, tarr, l, zmax, E);
        edge_sum_kernel<<<edge_blocks, 256, 0, stream>>>(xin, ei, tarr, l, zmax, den, num, E);
        combine_kernel<<<rowwave_blocks, 256, 0, stream>>>(xin, num, den, mscale, l, hp16, N);
        // GENConv MLP: u = relu(LN(hpre @ W1 + b1));  h = [h +] u @ W2 + b2
        gemm_rowblock<2, 8, 1><<<gemm_blocks, 256, 0, stream>>>(
            hp16, W1s + (size_t)l * HID * HID2, b1 + (size_t)l * HID2,
            g1 + (size_t)l * HID2, bb1 + (size_t)l * HID2, nullptr, u16, N, 0);
        gemm_rowblock<4, 4, 0><<<gemm_blocks, 256, 0, stream>>>(
            u16, W2s + (size_t)l * HID2 * HID, b2 + (size_t)l * HID,
            nullptr, nullptr, h, nullptr, N, (l > 0) ? 1 : 0);
    }

    // final: out = relu(LN(h, nrm[0])) @ lin_W + lin_b
    ln_relu_kernel<<<rowwave_blocks, 256, 0, stream>>>(h, ng, nb, r32, r16, N);
    gemm_rowblock<2, 7, 0><<<gemm_blocks, 256, 0, stream>>>(
        r16, linWs, linb, nullptr, nullptr, out, nullptr, N, 0);
}
